// NtXentLoss_65867618451797
// MI455X (gfx1250) — compile-verified, matrix operand order checked by reading, not run on hardware
//
#include <hip/hip_runtime.h>
#include <hip/hip_bf16.h>

typedef __attribute__((ext_vector_type(16))) _Float16 v16h;
typedef __attribute__((ext_vector_type(8)))  _Float16 h8;
typedef __attribute__((ext_vector_type(8)))  float    v8f;

#define DDIM   256
#define NROWS  8192
#define HALFB  4096
#define NCHUNK 8
#define JCHUNK (NROWS / NCHUNK)   // 1024 columns per chunk
#define INV_T  2.0f               // 1 / 0.5
// exp(s/T) = exp2(s * (1/T) * log2(e))
#define EXP2_SCALE 2.885390081777927f

// ---------------------------------------------------------------------------
// Kernel 1: L2-normalize rows of [features; transformed] -> f16 X in workspace
// One wave (32 lanes) per row; each lane owns 8 contiguous floats.
// ---------------------------------------------------------------------------
__global__ __launch_bounds__(32) void ntxent_norm(const float* __restrict__ feat,
                                                  const float* __restrict__ tfeat,
                                                  _Float16* __restrict__ Xh) {
    const int row  = blockIdx.x;                      // 0..8191
    const int lane = threadIdx.x;                     // 0..31
    const float* src = (row < HALFB) ? (feat + (size_t)row * DDIM)
                                     : (tfeat + (size_t)(row - HALFB) * DDIM);
    const float4 x0 = ((const float4*)src)[lane * 2 + 0];
    const float4 x1 = ((const float4*)src)[lane * 2 + 1];
    float ss = x0.x * x0.x + x0.y * x0.y + x0.z * x0.z + x0.w * x0.w
             + x1.x * x1.x + x1.y * x1.y + x1.z * x1.z + x1.w * x1.w;
#pragma unroll
    for (int m = 1; m < 32; m <<= 1) ss += __shfl_xor(ss, m, 32);
    const float inv = 1.0f / fmaxf(sqrtf(ss), 1e-12f);

    h8 out;
    out[0] = (_Float16)(x0.x * inv); out[1] = (_Float16)(x0.y * inv);
    out[2] = (_Float16)(x0.z * inv); out[3] = (_Float16)(x0.w * inv);
    out[4] = (_Float16)(x1.x * inv); out[5] = (_Float16)(x1.y * inv);
    out[6] = (_Float16)(x1.z * inv); out[7] = (_Float16)(x1.w * inv);
    *(h8*)(Xh + (size_t)row * DDIM + lane * 8) = out;
}

// ---------------------------------------------------------------------------
// Kernel 2: fused tile-GEMM (WMMA f16->f32) + exp + denominator/positive
// accumulation.  Never materializes the 8192x8192 sim matrix.
//
// grid = 64 M-groups x 8 J-chunks.  Block = 256 threads = 8 waves.
// Wave w owns M-tile rows [mGroup*128 + w*16, +16).  The 16-row B tile for
// each J step is staged once in (double-buffered) LDS, reused by all 8 waves.
// ---------------------------------------------------------------------------
__device__ __forceinline__ void stage_tile(const _Float16* __restrict__ Xh,
                                           _Float16* __restrict__ buf,
                                           int j, int tid) {
    const int r  = tid >> 4;          // 0..15  (row of B tile)
    const int cc = (tid & 15) * 16;   // 16-half chunk within row
    const h8* src = (const h8*)(Xh + (size_t)(j + r) * DDIM + cc);
    h8* dst = (h8*)(buf + r * DDIM + cc);
    dst[0] = src[0];
    dst[1] = src[1];
}

__global__ __launch_bounds__(256) void ntxent_main(const _Float16* __restrict__ Xh,
                                                   float* __restrict__ denomPart,
                                                   float* __restrict__ posPart) {
    const int tid     = threadIdx.x;
    const int wave    = tid >> 5;
    const int lane    = tid & 31;
    const int mGroup  = blockIdx.x & 63;      // 64 groups of 128 rows
    const int chunk   = blockIdx.x >> 6;      // 0..7 column chunk
    const int rowBase = mGroup * 128 + wave * 16;
    const int jBase   = chunk * JCHUNK;

    const int laneHalf = lane >> 4;           // 0: lanes 0-15, 1: lanes 16-31
    const int lane15   = lane & 15;
    const int mOfSlot0 = laneHalf * 8;        // row offset of C-slot v in this half

    __shared__ _Float16 Bt[2][16 * DDIM];     // 2 x 8 KB double-buffered B tile

    // --- Preload A fragments for all K (8 x 32-K chunks), per ISA 16-bit A
    // layout: lanes 0-15 hold K {0..7, 16..23}; lanes 16-31 hold {8..15, 24..31}.
    v16h a[8];
    {
        const _Float16* arow = Xh + (size_t)(rowBase + lane15) * DDIM;
#pragma unroll
        for (int kc = 0; kc < 8; ++kc) {
            const int kb = kc * 32 + laneHalf * 8;
            const h8 lo = *(const h8*)(arow + kb);
            const h8 hi = *(const h8*)(arow + kb + 16);
            v16h av;
#pragma unroll
            for (int i = 0; i < 8; ++i) { av[i] = lo[i]; av[i + 8] = hi[i]; }
            a[kc] = av;
        }
    }

    float denom[8], pos[8];
#pragma unroll
    for (int v = 0; v < 8; ++v) { denom[v] = 0.0f; pos[v] = 0.0f; }

    // Prime first buffer.
    stage_tile(Xh, Bt[0], jBase, tid);
    __syncthreads();

    int cur = 0;
    for (int j = jBase; j < jBase + JCHUNK; j += 16) {
        // Stage next tile into the other buffer while computing this one.
        if (j + 16 < jBase + JCHUNK) stage_tile(Xh, Bt[cur ^ 1], j + 16, tid);

        // Batch-load all 8 B fragments so ds_loads overlap the WMMA chain.
        // B column n = X row (j+n); same per-lane K swizzle as A.
        v16h b[8];
        {
            const _Float16* brow = Bt[cur] + lane15 * DDIM;
#pragma unroll
            for (int kc = 0; kc < 8; ++kc) {
                const int kb = kc * 32 + laneHalf * 8;
                const h8 lo = *(const h8*)(brow + kb);
                const h8 hi = *(const h8*)(brow + kb + 16);
                v16h bv;
#pragma unroll
                for (int i = 0; i < 8; ++i) { bv[i] = lo[i]; bv[i + 8] = hi[i]; }
                b[kc] = bv;
            }
        }

        // Two independent accumulator chains, interleaved.
        v8f c0 = {}, c1 = {};
#pragma unroll
        for (int kc = 0; kc < 4; ++kc) {
            c0 = __builtin_amdgcn_wmma_f32_16x16x32_f16(
                     false, a[kc],     false, b[kc],     (short)0, c0, false, false);
            c1 = __builtin_amdgcn_wmma_f32_16x16x32_f16(
                     false, a[kc + 4], false, b[kc + 4], (short)0, c1, false, false);
        }

        // Epilogue.  Diagonal exists only in tile j == rowBase; positives only
        // in tile j == rowBase ^ 4096 (both wave-uniform scalar branches).
        if (j == rowBase) {                       // tile containing the diagonal
#pragma unroll
            for (int v = 0; v < 8; ++v) {
                const float s = c0[v] + c1[v];
                const bool diag = (lane15 == v + mOfSlot0);
                denom[v] += diag ? 1.0f
                                 : __builtin_amdgcn_exp2f(s * EXP2_SCALE);
            }
        } else {
#pragma unroll
            for (int v = 0; v < 8; ++v) {
                const float s = c0[v] + c1[v];
                denom[v] += __builtin_amdgcn_exp2f(s * EXP2_SCALE);
            }
        }
        if (j == (rowBase ^ HALFB)) {             // tile containing the positives
#pragma unroll
            for (int v = 0; v < 8; ++v) {
                const float s = c0[v] + c1[v];
                pos[v] += (lane15 == v + mOfSlot0) ? s : 0.0f;
            }
        }

        __syncthreads();    // staging done AND all reads of Bt[cur] done
        cur ^= 1;
    }

    // Reduce across the 16-lane half that shares each row.
#pragma unroll
    for (int v = 0; v < 8; ++v) {
        float d = denom[v], p = pos[v];
#pragma unroll
        for (int m = 1; m < 16; m <<= 1) {
            d += __shfl_xor(d, m, 32);
            p += __shfl_xor(p, m, 32);
        }
        denom[v] = d; pos[v] = p;
    }
    if (lane15 == 0) {  // lane 0 -> rows +0..7 ; lane 16 -> rows +8..15
        const int rbase = rowBase + mOfSlot0;
#pragma unroll
        for (int v = 0; v < 8; ++v) {
            denomPart[(size_t)chunk * NROWS + rbase + v] = denom[v];
            posPart  [(size_t)chunk * NROWS + rbase + v] = pos[v];
        }
    }
}

// ---------------------------------------------------------------------------
// Kernel 3: per-row loss = log(denom) - pos/T, block partial sums (32 blocks).
// ---------------------------------------------------------------------------
__global__ __launch_bounds__(256) void ntxent_rowloss(const float* __restrict__ dp,
                                                      const float* __restrict__ pp,
                                                      float* __restrict__ bsum) {
    const int r = blockIdx.x * 256 + threadIdx.x;
    float denom = 0.0f, pos = 0.0f;
#pragma unroll
    for (int c = 0; c < NCHUNK; ++c) {
        denom += dp[(size_t)c * NROWS + r];
        pos   += pp[(size_t)c * NROWS + r];
    }
    float loss = __logf(denom) - pos * INV_T;
#pragma unroll
    for (int m = 1; m < 32; m <<= 1) loss += __shfl_xor(loss, m, 32);
    __shared__ float red[8];
    if ((threadIdx.x & 31) == 0) red[threadIdx.x >> 5] = loss;
    __syncthreads();
    if (threadIdx.x < 8) {
        float v = red[threadIdx.x];
#pragma unroll
        for (int m = 1; m < 8; m <<= 1) v += __shfl_xor(v, m, 32);
        if (threadIdx.x == 0) bsum[blockIdx.x] = v;
    }
}

// ---------------------------------------------------------------------------
// Kernel 4: final mean over 32 block partials.
// ---------------------------------------------------------------------------
__global__ __launch_bounds__(32) void ntxent_final(const float* __restrict__ bsum,
                                                   float* __restrict__ out) {
    float v = bsum[threadIdx.x];
#pragma unroll
    for (int m = 1; m < 32; m <<= 1) v += __shfl_xor(v, m, 32);
    if (threadIdx.x == 0) out[0] = v * (1.0f / (float)NROWS);
}

// ---------------------------------------------------------------------------
extern "C" void kernel_launch(void* const* d_in, const int* in_sizes, int n_in,
                              void* d_out, int out_size, void* d_ws, size_t ws_size,
                              hipStream_t stream) {
    const float* feat  = (const float*)d_in[0];
    const float* tfeat = (const float*)d_in[1];
    float* out = (float*)d_out;

    char* ws = (char*)d_ws;
    _Float16* Xh = (_Float16*)ws;                                  // 4 MB
    float* dp   = (float*)(ws + (size_t)NROWS * DDIM * sizeof(_Float16));
    float* pp   = dp + (size_t)NCHUNK * NROWS;                     // 256 KB each
    float* bsum = pp + (size_t)NCHUNK * NROWS;                     // 128 B

    ntxent_norm   <<<NROWS, 32, 0, stream>>>(feat, tfeat, Xh);
    ntxent_main   <<<64 * NCHUNK, 256, 0, stream>>>(Xh, dp, pp);
    ntxent_rowloss<<<NROWS / 256, 256, 0, stream>>>(dp, pp, bsum);
    ntxent_final  <<<1, 32, 0, stream>>>(bsum, out);
}
